// CorrEncoderSimple_3762391351428
// MI455X (gfx1250) — matrix operand order repaired and checked by reading, not compiled, not run on hardware
//
#include <hip/hip_runtime.h>

// FlowNet-style dilated local correlation (MD=10, DIL=2, 441 displacements)
// fused leaky-relu + max over (prev,next) pairs, via f16 WMMA Gram blocks.

typedef __attribute__((ext_vector_type(16))) _Float16 v16h;
typedef __attribute__((ext_vector_type(8)))  float    v8f;
typedef __attribute__((ext_vector_type(2)))  float    v2f;

#define MD 10
#define ND 21               // 2*MD+1 displacements per axis
#define NEG_SLOPE 0.1f

#define WMMA_F16(A, B, Cacc) \
    __builtin_amdgcn_wmma_f32_16x16x32_f16(false, (A), false, (B), (short)0, (Cacc), false, false)

// One wave (32 threads) per (b, y, dyi, 32-pixel x-block); both parities in
// one k-loop sharing b64 loads (parity pair = adjacent floats).
template<int C, int H, int W>
__global__ __launch_bounds__(32)
void corr_level_wmma(const float* __restrict__ prv,
                     const float* __restrict__ cur,
                     const float* __restrict__ nxt,
                     float* __restrict__ out)
{
    constexpr int HW = H * W;

    const int lane = threadIdx.x;     // wave32
    const int n    = lane & 15;
    const int hi   = lane >> 4;

    const int bx   = blockIdx.x;
    const int dyi  = bx % ND;
    const int xb   = bx / ND;
    const int x0   = xb * 32;
    const int y    = blockIdx.y;
    const int b    = blockIdx.z;

    float* ob = out + (((size_t)b * (ND * ND) + (size_t)dyi * ND) * H + (size_t)y) * W;

    const int y2 = y + (2 * dyi - 2 * MD);
    if (y2 < 0 || y2 >= H) {
        // whole dy-row zero-padded: corr = 0, leaky(0) = 0, max(0,0) = 0
        const int x = x0 + lane;
        if (x < W) {
            #pragma unroll
            for (int j = 0; j < ND; ++j)
                ob[j * HW + x] = 0.0f;
        }
        return;
    }

    __shared__ float tile[ND * 33];   // 21 dx rows x 32 px, pad 33 (no bank conflicts)

    const float* pc = cur + (size_t)b * C * HW + (size_t)y2 * W;  // + c*HW + x
    const float* pp = prv + (size_t)b * C * HW + (size_t)y  * W;
    const float* pn = nxt + (size_t)b * C * HW + (size_t)y  * W;

    // Clamped even base pixel for A (garbage rows are masked at the store).
    int xeA = x0 + 2 * n;
    if (xeA > W - 2) xeA = W - 2;
    const int aBase = xeA + hi * (16 * HW);

    // Clamped even base pixels for the 3 B tiles; out-of-range columns are
    // zeroed at scatter time (whole dot product is zero there), so the inner
    // loop stays branch-free.
    int  bBase[3];
    bool bOK[3];
    #pragma unroll
    for (int q = 0; q < 3; ++q) {
        int xo = x0 - 2 * MD + 32 * q + 2 * n;          // even
        bOK[q] = (xo >= 0) && (xo < W);                 // covers both parities
        xo = xo < 0 ? 0 : (xo > W - 2 ? W - 2 : xo);
        bBase[q] = xo + hi * (16 * HW);
    }

    const v8f z = {0.f, 0.f, 0.f, 0.f, 0.f, 0.f, 0.f, 0.f};
    v8f accP[2][3] = {{z, z, z}, {z, z, z}};   // [parity][q] Gram(prev, cur)
    v8f accN[2][3] = {{z, z, z}, {z, z, z}};   // [parity][q] Gram(next, cur)

    for (int k0 = 0; k0 < C; k0 += 32) {
        const float* ppk = pp + k0 * HW;
        const float* pnk = pn + k0 * HW;
        const float* pck = pc + k0 * HW;

        // A fragments (16x32 f16): elem e -> K = e + 16*hi, row M = n.
        v16h aP0, aP1, aN0, aN1;
        #pragma unroll
        for (int e = 0; e < 16; ++e) {
            const int off = e * HW + aBase;  // imm-foldable (HW constexpr)
            const v2f vp = *(const v2f*)(ppk + off);     // parity pair in one b64
            const v2f vn = *(const v2f*)(pnk + off);
            aP0[e] = (_Float16)vp.x;  aP1[e] = (_Float16)vp.y;
            aN0[e] = (_Float16)vn.x;  aN1[e] = (_Float16)vn.y;
        }

        #pragma unroll
        for (int q = 0; q < 3; ++q) {
            // B fragment (32x16 f16): elem e -> K = e + 16*hi, col N = n.
            v16h b0, b1;
            #pragma unroll
            for (int e = 0; e < 16; ++e) {
                const v2f vb = *(const v2f*)(pck + e * HW + bBase[q]);
                b0[e] = (_Float16)vb.x;  b1[e] = (_Float16)vb.y;
            }
            accP[0][q] = WMMA_F16(aP0, b0, accP[0][q]);
            accN[0][q] = WMMA_F16(aN0, b0, accN[0][q]);
            accP[1][q] = WMMA_F16(aP1, b1, accP[1][q]);
            accN[1][q] = WMMA_F16(aN1, b1, accN[1][q]);
        }
    }

    // Scatter banded Gram entries -> LDS tile.
    // D layout: elem r holds (M = r + 8*hi, N = n); entry is corr for pixel
    // x = x0 + p + 2M, displacement index j = 16q + n - M.
    #pragma unroll
    for (int p = 0; p < 2; ++p) {
        #pragma unroll
        for (int q = 0; q < 3; ++q) {
            #pragma unroll
            for (int r = 0; r < 8; ++r) {
                const int M = r + 8 * hi;
                const int j = 16 * q + n - M;
                if (j >= 0 && j < ND) {
                    float v = fmaxf(accP[p][q][r], accN[p][q][r]); // max∘leaky == leaky∘max
                    v = (v > 0.f) ? v : (NEG_SLOPE * v);
                    tile[j * 33 + p + 2 * M] = bOK[q] ? v : 0.0f;
                }
            }
        }
    }

    __syncthreads();

    // Coalesced store: 21 rows of 32 consecutive pixels
    const int x = x0 + lane;
    if (x < W) {
        #pragma unroll
        for (int j = 0; j < ND; ++j)
            ob[j * HW + x] = tile[j * 33 + lane];
    }
}

template<int C, int H, int W>
static void launch_level(const float* prv, const float* cur, const float* nxt,
                         float* out, hipStream_t stream) {
    const int xblocks = (W + 31) / 32;
    dim3 grid(xblocks * ND, H, 4);
    corr_level_wmma<C, H, W><<<grid, dim3(32), 0, stream>>>(prv, cur, nxt, out);
}

extern "C" void kernel_launch(void* const* d_in, const int* in_sizes, int n_in,
                              void* d_out, int out_size, void* d_ws, size_t ws_size,
                              hipStream_t stream) {
    (void)in_sizes; (void)n_in; (void)out_size; (void)d_ws; (void)ws_size;

    const float* prv0 = (const float*)d_in[0];
    const float* prv1 = (const float*)d_in[1];
    const float* prv2 = (const float*)d_in[2];
    const float* prv3 = (const float*)d_in[3];
    const float* cur0 = (const float*)d_in[4];
    const float* cur1 = (const float*)d_in[5];
    const float* cur2 = (const float*)d_in[6];
    const float* cur3 = (const float*)d_in[7];
    const float* nxt0 = (const float*)d_in[8];
    const float* nxt1 = (const float*)d_in[9];
    const float* nxt2 = (const float*)d_in[10];
    const float* nxt3 = (const float*)d_in[11];

    float* o = (float*)d_out;
    launch_level< 64, 96, 320>(prv0, cur0, nxt0, o, stream);
    o += (size_t)4 * ND * ND * 96 * 320;
    launch_level< 64, 48, 160>(prv1, cur1, nxt1, o, stream);
    o += (size_t)4 * ND * ND * 48 * 160;
    launch_level<128, 24,  80>(prv2, cur2, nxt2, o, stream);
    o += (size_t)4 * ND * ND * 24 * 80;
    launch_level<256, 12,  40>(prv3, cur3, nxt3, o, stream);
}
